// BlockMoE_79353815761474
// MI455X (gfx1250) — matrix operand order repaired and checked by reading, not compiled, not run on hardware
//
#include <hip/hip_runtime.h>
#include <hip/hip_bf16.h>

#define DEV __device__ __forceinline__

typedef __attribute__((ext_vector_type(16))) __bf16 v16bf;
typedef __attribute__((ext_vector_type(8)))  __bf16 v8bf;
typedef __attribute__((ext_vector_type(8)))  float  v8f;
typedef __attribute__((ext_vector_type(4)))  int    v4i;

// Problem constants (from reference setup)
#define BB   2
#define TT   2048
#define CC   1024
#define NH   16
#define NKV  8
#define HD   64
#define EE   8
#define HH   2048
#define NN   (BB*TT)      // 4096 tokens
#define CAP  2048         // 2*N*TOP_K/E
#define VTP  (TT+32)      // padded T stride for transposed V
#define RMS_EPS 1.1920929e-7f

// ----- CDNA5 async global->LDS path (guarded; falls back to reg staging) -----
#if defined(__has_builtin)
#if __has_builtin(__builtin_amdgcn_global_load_async_to_lds_b128) && \
    __has_builtin(__builtin_amdgcn_s_wait_asynccnt)
#define HAVE_ASYNC 1
#endif
#endif
#ifndef HAVE_ASYNC
#define HAVE_ASYNC 0
#endif

typedef __attribute__((address_space(1))) void as1v;
typedef __attribute__((address_space(3))) void as3v;
typedef __attribute__((address_space(1))) v4i  as1_v4i;
typedef __attribute__((address_space(3))) v4i  as3_v4i;

DEV void cp16(const __bf16* g, __bf16* l) {
#if HAVE_ASYNC
  __builtin_amdgcn_global_load_async_to_lds_b128(
      (as1_v4i*)(as1v*)g, (as3_v4i*)(as3v*)l, 0, 0);
#else
  *(v8bf*)l = *(const v8bf*)g;
#endif
}

// ---------------- fragment helpers (CDNA5 WMMA layouts) ----------------
// A-matrix 16x32 bf16: lane<16 -> K {0..7, 16..23}; lane>=16 -> K {8..15, 24..31}
DEV v16bf ldsplit(const __bf16* p) {
  v8bf lo = *(const v8bf*)p;
  v8bf hi = *(const v8bf*)(p + 16);
  return __builtin_shufflevector(lo, hi, 0,1,2,3,4,5,6,7,8,9,10,11,12,13,14,15);
}
// B-matrix 32x16 bf16: lane holds column n=lane&15; lane<16 -> K {0..7, 16..23}; lane>=16 -> K {8..15, 24..31}
DEV v16bf ld16(const __bf16* p) {
  v8bf lo = *(const v8bf*)p;
  v8bf hi = *(const v8bf*)(p + 16);
  return __builtin_shufflevector(lo, hi, 0,1,2,3,4,5,6,7,8,9,10,11,12,13,14,15);
}
DEV v8f wmma_bf(v16bf a, v16bf b, v8f c) {
  return __builtin_amdgcn_wmma_f32_16x16x32_bf16(false, a, false, b, (short)0, c, false, false);
}
DEV float wredsum16(float v) {
  #pragma unroll
  for (int off = 1; off < 16; off <<= 1) v += __shfl_xor(v, off, 32);
  return v;
}
DEV float wredmax16(float v) {
  #pragma unroll
  for (int off = 1; off < 16; off <<= 1) v = fmaxf(v, __shfl_xor(v, off, 32));
  return v;
}
DEV float wredsum32(float v) {
  #pragma unroll
  for (int off = 1; off < 32; off <<= 1) v += __shfl_xor(v, off, 32);
  return v;
}

// ---------------- elementwise kernels ----------------
__global__ void cvt_bf16_kernel(const float* __restrict__ s, __bf16* __restrict__ d, long n) {
  long i = (long)blockIdx.x * 256 + threadIdx.x;
  if (i < n) d[i] = (__bf16)s[i];
}

// RMSNorm over last dim (1024): writes f32 and bf16 copies. One block per row.
__global__ __launch_bounds__(256)
void rmsnorm_kernel(const float* __restrict__ x, float* __restrict__ xn,
                    __bf16* __restrict__ xnb) {
  int row = blockIdx.x;
  const float4* p = (const float4*)(x + (size_t)row * CC);
  float4 v = p[threadIdx.x];
  float s = v.x*v.x + v.y*v.y + v.z*v.z + v.w*v.w;
  s = wredsum32(s);
  __shared__ float red[8];
  if ((threadIdx.x & 31) == 0) red[threadIdx.x >> 5] = s;
  __syncthreads();
  float tot = 0.f;
  #pragma unroll
  for (int i = 0; i < 8; i++) tot += red[i];
  float sc = rsqrtf(tot * (1.0f / CC) + RMS_EPS);
  float4 o = make_float4(v.x*sc, v.y*sc, v.z*sc, v.w*sc);
  ((float4*)(xn + (size_t)row * CC))[threadIdx.x] = o;
  union { __bf16 h[4]; unsigned long long u; } pk;
  pk.h[0] = (__bf16)o.x; pk.h[1] = (__bf16)o.y;
  pk.h[2] = (__bf16)o.z; pk.h[3] = (__bf16)o.w;
  *(unsigned long long*)(xnb + (size_t)row * CC + threadIdx.x * 4) = pk.u;
}

// RoPE + RMS for q/k rows of 64. 8 rows (one wave each) per 256-thread block.
__global__ __launch_bounds__(256)
void rope_rms_kernel(const float* __restrict__ src, const float* __restrict__ cosb,
                     const float* __restrict__ sinb, __bf16* __restrict__ dst,
                     int nh, int total) {
  int row = blockIdx.x * 8 + (threadIdx.x >> 5);
  int lane = threadIdx.x & 31;
  if (row >= total) return;
  int t = (row / nh) % TT;
  const float* s = src + (size_t)row * HD;
  float c = cosb[t * 32 + lane], sn = sinb[t * 32 + lane];
  float x1 = s[lane], x2 = s[lane + 32];
  float r1 =  x1 * c + x2 * sn;
  float r2 = -x1 * sn + x2 * c;
  float ss = wredsum32(r1 * r1 + r2 * r2);
  float sc = rsqrtf(ss * (1.0f / HD) + RMS_EPS);
  dst[(size_t)row * HD + lane]      = (__bf16)(r1 * sc);
  dst[(size_t)row * HD + lane + 32] = (__bf16)(r2 * sc);
}

// v = v + 2*sigmoid(xn[:, :32] @ gate_w[kv]^T) * ve ; store transposed (b,kv,hd,T) bf16
__global__ __launch_bounds__(256)
void vgate_kernel(const float* __restrict__ xn, const float* __restrict__ gw,
                  const float* __restrict__ vsrc, const float* __restrict__ ve,
                  __bf16* __restrict__ vt) {
  int n = blockIdx.x;            // token
  int b = n / TT, t = n % TT;
  int w = threadIdx.x >> 5;      // kv head
  int lane = threadIdx.x & 31;
  float g = xn[(size_t)n * CC + lane] * gw[w * 32 + lane];
  g = wredsum32(g);
  g = 2.0f / (1.0f + __expf(-g));
  #pragma unroll
  for (int rep = 0; rep < 2; rep++) {
    int d = lane + rep * 32;
    size_t si = (size_t)n * (NKV * HD) + w * HD + d;
    float val = vsrc[si] + g * ve[si];
    vt[(((size_t)b * NKV + w) * HD + d) * VTP + t] = (__bf16)val;
  }
}

// ---------------- WMMA GEMM: C[M,N] = A[M,K] * W[N,K]^T ----------------
// mode 0: Cf = AW^T   mode 1: Cf = AW^T + R   mode 2: Cb = bf16(relu(AW^T)^2)
// Block 256 threads = 8 waves (4 along M x 2 along N); wave tile 32x64;
// block tile 128x128. K-slices (128x32 of A and W) are staged into LDS with
// async DMA (double-buffered), fragments read back via ds_load_b128.
#define LDP 40   // padded LDS row stride (elements) to spread banks
__global__ __launch_bounds__(256)
void gemm_kernel(const __bf16* __restrict__ A, int lda, long long sA,
                 const __bf16* __restrict__ W, int ldw, long long sW,
                 float* __restrict__ Cf, __bf16* __restrict__ Cb, int ldc, long long sC,
                 const float* __restrict__ R, int ldr, int K, int mode) {
  __shared__ __bf16 As[2][128 * LDP];
  __shared__ __bf16 Bs[2][128 * LDP];
  int tid = threadIdx.x;
  int lane = tid & 31;
  int wv = tid >> 5;
  int wm = wv & 3;          // wave M index (0..3)
  int wn = wv >> 2;         // wave N index (0..1)
  int lc = lane & 15;
  int aoff = (lane >> 4) << 3;
  int boff = (lane >> 4) << 3;
  size_t z = blockIdx.z;
  const __bf16* ga = A + z * (size_t)sA + (size_t)(blockIdx.y * 128) * lda;
  const __bf16* gw = W + z * (size_t)sW + (size_t)(blockIdx.x * 128) * ldw;

  // two 16B chunks per matrix per thread cover the 128x32 tile
  int c0 = tid * 2;
  int r0 = c0 >> 2, col0 = (c0 & 3) << 3;
  int c1 = c0 + 1;
  int r1 = c1 >> 2, col1 = (c1 & 3) << 3;

  auto stage = [&](int buf, int k) {
    cp16(ga + (size_t)r0 * lda + k + col0, &As[buf][r0 * LDP + col0]);
    cp16(gw + (size_t)r0 * ldw + k + col0, &Bs[buf][r0 * LDP + col0]);
    cp16(ga + (size_t)r1 * lda + k + col1, &As[buf][r1 * LDP + col1]);
    cp16(gw + (size_t)r1 * ldw + k + col1, &Bs[buf][r1 * LDP + col1]);
  };

  v8f acc[2][4] = {};
  int nk = K >> 5;
  stage(0, 0);
  for (int kb = 0; kb < nk; ++kb) {
    int cur = kb & 1;
    if (kb + 1 < nk) {
      stage(cur ^ 1, (kb + 1) << 5);
#if HAVE_ASYNC
      __builtin_amdgcn_s_wait_asynccnt(4);   // current buffer's 4 DMAs done
#endif
    } else {
#if HAVE_ASYNC
      __builtin_amdgcn_s_wait_asynccnt(0);
#endif
    }
    __syncthreads();
    v16bf af[2], bfr[4];
    #pragma unroll
    for (int i = 0; i < 2; i++)
      af[i] = ldsplit(&As[cur][(wm * 32 + i * 16 + lc) * LDP + aoff]);
    #pragma unroll
    for (int j = 0; j < 4; j++)
      bfr[j] = ld16(&Bs[cur][(wn * 64 + j * 16 + lc) * LDP + boff]);
    #pragma unroll
    for (int i = 0; i < 2; i++)
      #pragma unroll
      for (int j = 0; j < 4; j++)
        acc[i][j] = wmma_bf(af[i], bfr[j], acc[i][j]);
    __syncthreads();
  }
  int hf8 = (lane >> 4) << 3;
  int mbase = blockIdx.y * 128 + wm * 32;
  int nbase = blockIdx.x * 128 + wn * 64;
  #pragma unroll
  for (int i = 0; i < 2; i++)
  #pragma unroll
  for (int j = 0; j < 4; j++) {
    #pragma unroll
    for (int r = 0; r < 8; r++) {
      int row = mbase + i * 16 + r + hf8;
      int col = nbase + j * 16 + lc;
      size_t idx = z * (size_t)sC + (size_t)row * ldc + col;
      float v = acc[i][j][r];
      if (mode == 0)      Cf[idx] = v;
      else if (mode == 1) Cf[idx] = v + R[(size_t)row * ldr + col];
      else { float h = v > 0.f ? v : 0.f; Cb[idx] = (__bf16)(h * h); }
    }
  }
}

// ---------------- flash attention, one wave per 16-query tile ----------------
__global__ __launch_bounds__(32)
void attn_kernel(const __bf16* __restrict__ qb, const __bf16* __restrict__ kb,
                 const __bf16* __restrict__ vt, __bf16* __restrict__ y,
                 const int* __restrict__ winp) {
  int lane = threadIdx.x;
  int q0 = blockIdx.x * 16;
  int h = blockIdx.y;
  int b = blockIdx.z;
  int kv = h >> 1;               // rep = NH/NKV = 2
  int win = *winp;
  int hf = lane >> 4;
  int lc = lane & 15;
  int aoff = hf << 3;
  int boff = hf << 3;

  const __bf16* qrow = qb + (((size_t)b * TT + (q0 + lc)) * NH + h) * HD;
  v16bf aq0 = ldsplit(qrow + aoff);
  v16bf aq1 = ldsplit(qrow + 32 + aoff);

  float ms[8], ls[8];
  #pragma unroll
  for (int r = 0; r < 8; r++) { ms[r] = -3.0e38f; ls[r] = 0.f; }
  v8f acc[4] = {};

  __shared__ float pl[16][32];

  int s_lo = q0 - win;
  if (s_lo < 0) s_lo = 0;
  s_lo &= ~31;

  for (int s0 = s_lo; s0 <= q0 + 15; s0 += 32) {
    v8f sc[2];
    #pragma unroll
    for (int kt = 0; kt < 2; kt++) {
      int key = s0 + kt * 16 + lc;
      if (key > TT - 1) key = TT - 1;
      const __bf16* krow = kb + (((size_t)b * TT + key) * NKV + kv) * HD;
      v16bf bk0 = ld16(krow + boff);
      v16bf bk1 = ld16(krow + 32 + boff);
      v8f z = {};
      z = wmma_bf(aq0, bk0, z);
      z = wmma_bf(aq1, bk1, z);
      sc[kt] = z;
    }
    // mask + scale + online softmax (rows in D-layout: m = r + hf*8, col n = lc)
    #pragma unroll
    for (int r = 0; r < 8; r++) {
      int i = q0 + r + hf * 8;
      int j0 = s0 + lc, j1 = s0 + 16 + lc;
      float s0v = (j0 <= i && i - j0 <= win) ? sc[0][r] * 0.125f : -1.0e30f;
      float s1v = (j1 <= i && i - j1 <= win) ? sc[1][r] * 0.125f : -1.0e30f;
      float mx = wredmax16(fmaxf(s0v, s1v));
      float mn = fmaxf(ms[r], mx);
      float scl = __expf(ms[r] - mn);
      float p0 = __expf(s0v - mn);
      float p1 = __expf(s1v - mn);
      float rs = wredsum16(p0 + p1);
      ls[r] = ls[r] * scl + rs;
      ms[r] = mn;
      #pragma unroll
      for (int j = 0; j < 4; j++) acc[j][r] *= scl;
      sc[0][r] = p0; sc[1][r] = p1;
    }
    // transpose P (D-layout -> A-layout) via LDS
    __syncthreads();
    #pragma unroll
    for (int r = 0; r < 8; r++) {
      int mrow = r + hf * 8;
      pl[mrow][lc] = sc[0][r];
      pl[mrow][16 + lc] = sc[1][r];
    }
    __syncthreads();
    v16bf pa;
    {
      const float* prow = pl[lc];
      #pragma unroll
      for (int i = 0; i < 8; i++) {
        pa[i]     = (__bf16)prow[aoff + i];
        pa[8 + i] = (__bf16)prow[aoff + 16 + i];
      }
    }
    // P @ V : V transposed layout (b,kv,hd,T), B-frag loads contiguous in T
    #pragma unroll
    for (int j = 0; j < 4; j++) {
      const __bf16* vrow = vt + (((size_t)b * NKV + kv) * HD + (j * 16 + lc)) * VTP + s0 + boff;
      v16bf bv = ld16(vrow);
      acc[j] = wmma_bf(pa, bv, acc[j]);
    }
  }
  // write y (B,T,C) bf16
  #pragma unroll
  for (int j = 0; j < 4; j++) {
    #pragma unroll
    for (int r = 0; r < 8; r++) {
      int m = q0 + r + hf * 8;
      int col = h * HD + j * 16 + lc;
      y[((size_t)b * TT + m) * CC + col] = (__bf16)(acc[j][r] / ls[r]);
    }
  }
}

// ---------------- router: logits, softmax, top-2 ----------------
__global__ __launch_bounds__(256)
void router_kernel(const float* __restrict__ xn2, const float* __restrict__ rw_w,
                   float* __restrict__ rw_out, int* __restrict__ tke,
                   float* __restrict__ tkw) {
  int n = blockIdx.x;
  int w = threadIdx.x >> 5;
  int lane = threadIdx.x & 31;
  const float* xr = xn2 + (size_t)n * CC;
  const float* wr = rw_w + (size_t)w * CC;
  float s = 0.f;
  for (int i = lane; i < CC; i += 32) s += xr[i] * wr[i];
  s = wredsum32(s);
  __shared__ float lg[EE];
  if (lane == 0) lg[w] = s;
  __syncthreads();
  if (threadIdx.x == 0) {
    float mx = lg[0];
    #pragma unroll
    for (int j = 1; j < EE; j++) mx = fmaxf(mx, lg[j]);
    float e[EE], sum = 0.f;
    #pragma unroll
    for (int j = 0; j < EE; j++) { e[j] = __expf(lg[j] - mx); sum += e[j]; }
    float inv = 1.0f / sum;
    #pragma unroll
    for (int j = 0; j < EE; j++) { e[j] *= inv; rw_out[(size_t)n * EE + j] = e[j]; }
    int i0 = 0;
    #pragma unroll
    for (int j = 1; j < EE; j++) if (e[j] > e[i0]) i0 = j;
    int i1 = (i0 == 0) ? 1 : 0;
    #pragma unroll
    for (int j = 0; j < EE; j++) if (j != i0 && e[j] > e[i1]) i1 = j;
    float iv = 1.0f / (e[i0] + e[i1] + 1e-10f);
    tke[2 * n] = i0;     tke[2 * n + 1] = i1;
    tkw[2 * n] = e[i0] * iv; tkw[2 * n + 1] = e[i1] * iv;
  }
}

// stable positions within each expert (flat order = token*2 + slot)
__global__ void pos_kernel(const int* __restrict__ tke, int* __restrict__ pos) {
  int e = threadIdx.x;
  if (e >= EE) return;
  int cnt = 0;
  for (int i = 0; i < NN * 2; i++)
    if (tke[i] == e) pos[i] = cnt++;
}

__global__ void fillslot_kernel(int* __restrict__ slot) {
  slot[blockIdx.x * 256 + threadIdx.x] = -1;
}

__global__ void scatter_kernel(const int* __restrict__ tke, const int* __restrict__ pos,
                               int* __restrict__ slot) {
  int i = blockIdx.x * 256 + threadIdx.x;   // < 2N
  int p = pos[i];
  if (p < CAP) slot[tke[i] * CAP + p] = i >> 1;
}

__global__ __launch_bounds__(256)
void gather_kernel(const int* __restrict__ slot, const __bf16* __restrict__ xn2b,
                   __bf16* __restrict__ bx) {
  int s = blockIdx.x;
  int tok = slot[s];
  unsigned long long* d = (unsigned long long*)(bx + (size_t)s * CC);
  if (tok >= 0) {
    const unsigned long long* sp = (const unsigned long long*)(xn2b + (size_t)tok * CC);
    d[threadIdx.x] = sp[threadIdx.x];   // 256 * 8B = 2048B = 1024 bf16
  } else {
    d[threadIdx.x] = 0ULL;
  }
}

__global__ __launch_bounds__(256)
void combine_kernel(const float* __restrict__ x2, const float* __restrict__ bo,
                    const int* __restrict__ tke, const int* __restrict__ pos,
                    const float* __restrict__ tkw, float* __restrict__ out) {
  int n = blockIdx.x;
  int e0 = tke[2 * n], e1 = tke[2 * n + 1];
  int p0 = pos[2 * n], p1 = pos[2 * n + 1];
  float w0 = (p0 < CAP) ? tkw[2 * n] : 0.f;
  float w1 = (p1 < CAP) ? tkw[2 * n + 1] : 0.f;
  const float* b0 = bo + ((size_t)e0 * CAP + (p0 < CAP ? p0 : CAP - 1)) * CC;
  const float* b1 = bo + ((size_t)e1 * CAP + (p1 < CAP ? p1 : CAP - 1)) * CC;
  for (int i = threadIdx.x; i < CC; i += 256) {
    out[(size_t)n * CC + i] = x2[(size_t)n * CC + i] + w0 * b0[i] + w1 * b1[i];
  }
}

// ---------------- host orchestration ----------------
extern "C" void kernel_launch(void* const* d_in, const int* in_sizes, int n_in,
                              void* d_out, int out_size, void* d_ws, size_t ws_size,
                              hipStream_t stream) {
  const float* x    = (const float*)d_in[0];
  const float* ve   = (const float*)d_in[1];
  const float* cosb = (const float*)d_in[2];
  const float* sinb = (const float*)d_in[3];
  const float* wq   = (const float*)d_in[4];
  const float* wk   = (const float*)d_in[5];
  const float* wv   = (const float*)d_in[6];
  const float* wp   = (const float*)d_in[7];
  const float* gww  = (const float*)d_in[8];
  const float* rww  = (const float*)d_in[9];
  const float* fcw  = (const float*)d_in[10];
  const float* pjw  = (const float*)d_in[11];
  const int*   winp = (const int*)d_in[12];

  float* out_x  = (float*)d_out;
  float* out_rw = out_x + (size_t)NN * CC;

  size_t off = 0;
  char* base = (char*)d_ws;
  auto alloc = [&](size_t bytes) -> char* {
    off = (off + 255) & ~(size_t)255;
    char* r = base + off;
    off += bytes;
    return r;
  };
  float*  xn_f   = (float*) alloc((size_t)NN * CC * 4);
  __bf16* xn_b   = (__bf16*)alloc((size_t)NN * CC * 2);
  __bf16* wq_b   = (__bf16*)alloc((size_t)CC * CC * 2);
  __bf16* wk_b   = (__bf16*)alloc((size_t)(NKV*HD) * CC * 2);
  __bf16* wv_b   = (__bf16*)alloc((size_t)(NKV*HD) * CC * 2);
  __bf16* wp_b   = (__bf16*)alloc((size_t)CC * CC * 2);
  __bf16* fcw_b  = (__bf16*)alloc((size_t)EE * HH * CC * 2);
  __bf16* pjw_b  = (__bf16*)alloc((size_t)EE * CC * HH * 2);
  float*  q_f    = (float*) alloc((size_t)NN * CC * 4);
  float*  k_f    = (float*) alloc((size_t)NN * (NKV*HD) * 4);
  float*  v_f    = (float*) alloc((size_t)NN * (NKV*HD) * 4);
  __bf16* q_b    = (__bf16*)alloc((size_t)NN * CC * 2);
  __bf16* k_b    = (__bf16*)alloc((size_t)NN * (NKV*HD) * 2);
  __bf16* vt_b   = (__bf16*)alloc((size_t)BB * NKV * HD * VTP * 2);
  __bf16* y_b    = (__bf16*)alloc((size_t)NN * CC * 2);
  float*  x2     = (float*) alloc((size_t)NN * CC * 4);
  float*  xn2_f  = (float*) alloc((size_t)NN * CC * 4);
  __bf16* xn2_b  = (__bf16*)alloc((size_t)NN * CC * 2);
  int*    tke    = (int*)   alloc((size_t)NN * 2 * 4);
  float*  tkw    = (float*) alloc((size_t)NN * 2 * 4);
  int*    posb   = (int*)   alloc((size_t)NN * 2 * 4);
  int*    slotb  = (int*)   alloc((size_t)EE * CAP * 4);
  __bf16* bx     = (__bf16*)alloc((size_t)EE * CAP * CC * 2);
  __bf16* h_b    = (__bf16*)alloc((size_t)EE * CAP * HH * 2);
  float*  bo     = (float*) alloc((size_t)EE * CAP * CC * 4);
  (void)ws_size; (void)in_sizes; (void)n_in; (void)out_size;

  auto cvt = [&](const float* s, __bf16* d, long n) {
    cvt_bf16_kernel<<<dim3((unsigned)((n + 255) / 256)), dim3(256), 0, stream>>>(s, d, n);
  };
  // 1. weights -> bf16
  cvt(wq,  wq_b,  (long)CC * CC);
  cvt(wk,  wk_b,  (long)(NKV*HD) * CC);
  cvt(wv,  wv_b,  (long)(NKV*HD) * CC);
  cvt(wp,  wp_b,  (long)CC * CC);
  cvt(fcw, fcw_b, (long)EE * HH * CC);
  cvt(pjw, pjw_b, (long)EE * CC * HH);

  // 2. RMSNorm(x)
  rmsnorm_kernel<<<dim3(NN), dim3(256), 0, stream>>>(x, xn_f, xn_b);

  // 3. QKV GEMMs (block tile 128x128)
  gemm_kernel<<<dim3(CC/128, NN/128, 1), dim3(256), 0, stream>>>(
      xn_b, CC, 0, wq_b, CC, 0, q_f, nullptr, CC, 0, nullptr, 0, CC, 0);
  gemm_kernel<<<dim3((NKV*HD)/128, NN/128, 1), dim3(256), 0, stream>>>(
      xn_b, CC, 0, wk_b, CC, 0, k_f, nullptr, NKV*HD, 0, nullptr, 0, CC, 0);
  gemm_kernel<<<dim3((NKV*HD)/128, NN/128, 1), dim3(256), 0, stream>>>(
      xn_b, CC, 0, wv_b, CC, 0, v_f, nullptr, NKV*HD, 0, nullptr, 0, CC, 0);

  // 4. RoPE + RMS on q, k ; gate + transpose on v
  rope_rms_kernel<<<dim3(NN*NH/8), dim3(256), 0, stream>>>(q_f, cosb, sinb, q_b, NH, NN*NH);
  rope_rms_kernel<<<dim3(NN*NKV/8), dim3(256), 0, stream>>>(k_f, cosb, sinb, k_b, NKV, NN*NKV);
  vgate_kernel<<<dim3(NN), dim3(256), 0, stream>>>(xn_f, gww, v_f, ve, vt_b);

  // 5. attention
  attn_kernel<<<dim3(TT/16, NH, BB), dim3(32), 0, stream>>>(q_b, k_b, vt_b, y_b, winp);

  // 6. out-proj + residual: x2 = x + y @ wp^T
  gemm_kernel<<<dim3(CC/128, NN/128, 1), dim3(256), 0, stream>>>(
      y_b, CC, 0, wp_b, CC, 0, x2, nullptr, CC, 0, x, CC, CC, 1);

  // 7. RMSNorm(x2)
  rmsnorm_kernel<<<dim3(NN), dim3(256), 0, stream>>>(x2, xn2_f, xn2_b);

  // 8. router + top-2 (also writes rw output)
  router_kernel<<<dim3(NN), dim3(256), 0, stream>>>(xn2_f, rww, out_rw, tke, tkw);

  // 9. stable expert positions, slots, gather
  pos_kernel<<<dim3(1), dim3(32), 0, stream>>>(tke, posb);
  fillslot_kernel<<<dim3(EE*CAP/256), dim3(256), 0, stream>>>(slotb);
  scatter_kernel<<<dim3(NN*2/256), dim3(256), 0, stream>>>(tke, posb, slotb);
  gather_kernel<<<dim3(EE*CAP), dim3(256), 0, stream>>>(slotb, xn2_b, bx);

  // 10. expert GEMMs (batched over experts via grid.z)
  gemm_kernel<<<dim3(HH/128, CAP/128, EE), dim3(256), 0, stream>>>(
      bx, CC, (long long)CAP*CC, fcw_b, CC, (long long)HH*CC,
      nullptr, h_b, HH, (long long)CAP*HH, nullptr, 0, CC, 2);
  gemm_kernel<<<dim3(CC/128, CAP/128, EE), dim3(256), 0, stream>>>(
      h_b, HH, (long long)CAP*HH, pjw_b, HH, (long long)CC*HH,
      bo, nullptr, CC, (long long)CAP*CC, nullptr, 0, HH, 0);

  // 11. combine + residual -> output
  combine_kernel<<<dim3(NN), dim3(256), 0, stream>>>(x2, bo, tke, posb, tkw, out_x);
}
